// Het_GNN_11458972746417
// MI455X (gfx1250) — matrix operand order repaired and checked by reading, not compiled, not run on hardware
//
#include <hip/hip_runtime.h>
#include <hip/hip_bf16.h>
#include <cmath>

// ---------------------------------------------------------------------------
// Het_GNN forward for MI455X (gfx1250, wave32, WMMA).
// - Generic GEMM: v_wmma_f32_16x16x32_bf16, bf16 staging via v_cvt_pk_bf16_f32,
//   b128 global loads, b64 LDS stores, register double-buffered prefetch.
// - Fused BiLSTM: both gate GEMMs + sigmoid/tanh epilogue in one kernel; the
//   [32 x 800] fp32 gate pre-activations live entirely in LDS (~120 KB/WGP),
//   eliminating ~1.2 GB of HBM round-trip traffic vs. materializing Z.
// ---------------------------------------------------------------------------

typedef __bf16 bf16x16 __attribute__((ext_vector_type(16)));
typedef __bf16 bf16x4v __attribute__((ext_vector_type(4)));
typedef float  f32x8   __attribute__((ext_vector_type(8)));
typedef float  f32x4v  __attribute__((ext_vector_type(4)));
typedef unsigned int u32x8 __attribute__((ext_vector_type(8)));

#define TBM 128   // GEMM block tile M
#define TBN 64    // block tile N
#define TBK 32    // K chunk (WMMA bf16 K)
#define LDT 40    // padded LDS row stride in ushorts (80B rows)

#define FBM  32   // fused-BiLSTM M tile
#define KMAX 224  // max K for fused kernel (200 rounded up to 32)
#define FLDA 232  // fused A row stride (ushorts)
#define ZLD  804  // fused z-buffer row stride (floats)

// float4 -> 4x bf16 (packed) via two v_cvt_pk_bf16_f32
__device__ __forceinline__ uint2 cvt4(const float4& f) {
  f32x4v v = { f.x, f.y, f.z, f.w };
  bf16x4v h = __builtin_convertvector(v, bf16x4v);
  return __builtin_bit_cast(uint2, h);
}

// branch-free guarded scalar load: clamp address, select 0 (no exec-mask loop)
__device__ __forceinline__ float ldg_guard(const float* __restrict__ P, int ld,
                                           int r, int c, int Mr, int Kc) {
  int rr = r < Mr ? r : Mr - 1;
  int cc = c < Kc ? c : Kc - 1;
  float v = P[(size_t)rr * ld + cc];
  return (r < Mr && c < Kc) ? v : 0.0f;
}

// A tile: TBM x TBK floats = 1024 float4 chunks; 4 per thread (256 thr)
__device__ __forceinline__ void fetchA(float4* r, const float* __restrict__ X,
                                       int ldx, int m0, int k0, int M, int K,
                                       int tid) {
  const bool fast = (m0 + TBM <= M) && (k0 + TBK <= K) && ((ldx & 3) == 0);
  if (fast) {
#pragma unroll
    for (int i = 0; i < 4; ++i) {
      int ch = tid + 256 * i;
      int row = ch >> 3, c4 = (ch & 7) << 2;
      r[i] = *(const float4*)(X + (size_t)(m0 + row) * ldx + k0 + c4);
    }
  } else {
#pragma unroll
    for (int i = 0; i < 4; ++i) {
      int ch = tid + 256 * i;
      int row = m0 + (ch >> 3), c = k0 + ((ch & 7) << 2);
      r[i].x = ldg_guard(X, ldx, row, c + 0, M, K);
      r[i].y = ldg_guard(X, ldx, row, c + 1, M, K);
      r[i].z = ldg_guard(X, ldx, row, c + 2, M, K);
      r[i].w = ldg_guard(X, ldx, row, c + 3, M, K);
    }
  }
}

// B tile (= W rows): TBN x TBK floats = 512 float4 chunks; 2 per thread
__device__ __forceinline__ void fetchB(float4* r, const float* __restrict__ W,
                                       int ldw, int n0, int k0, int N, int K,
                                       int tid) {
  const bool fast = (n0 + TBN <= N) && (k0 + TBK <= K) && ((ldw & 3) == 0);
  if (fast) {
#pragma unroll
    for (int i = 0; i < 2; ++i) {
      int ch = tid + 256 * i;
      int row = ch >> 3, c4 = (ch & 7) << 2;
      r[i] = *(const float4*)(W + (size_t)(n0 + row) * ldw + k0 + c4);
    }
  } else {
#pragma unroll
    for (int i = 0; i < 2; ++i) {
      int ch = tid + 256 * i;
      int row = n0 + (ch >> 3), c = k0 + ((ch & 7) << 2);
      r[i].x = ldg_guard(W, ldw, row, c + 0, N, K);
      r[i].y = ldg_guard(W, ldw, row, c + 1, N, K);
      r[i].z = ldg_guard(W, ldw, row, c + 2, N, K);
      r[i].w = ldg_guard(W, ldw, row, c + 3, N, K);
    }
  }
}

__device__ __forceinline__ void storeA(unsigned short* Als, const float4* r,
                                       int tid) {
#pragma unroll
  for (int i = 0; i < 4; ++i) {
    int ch = tid + 256 * i;
    int row = ch >> 3, c4 = (ch & 7) << 2;
    *(uint2*)&Als[row * LDT + c4] = cvt4(r[i]);  // 8B-aligned ds_store_b64
  }
}
__device__ __forceinline__ void storeB(unsigned short* Bls, const float4* r,
                                       int tid) {
#pragma unroll
  for (int i = 0; i < 2; ++i) {
    int ch = tid + 256 * i;
    int row = ch >> 3, c4 = (ch & 7) << 2;
    *(uint2*)&Bls[row * LDT + c4] = cvt4(r[i]);
  }
}

// A fragment (16x32 bf16): lane = {row = lane&15, khalf = lane>>4}.
// Lane holds contiguous K in {kh*16 .. kh*16+15} (matches B fragment).
__device__ __forceinline__ bf16x16 load_fragA(const unsigned short* rowbase, int kh) {
  const unsigned int* p = (const unsigned int*)rowbase;
  u32x8 u;
  int o = kh * 8;
#pragma unroll
  for (int i = 0; i < 8; ++i) u[i] = p[o + i];
  return __builtin_bit_cast(bf16x16, u);
}

// B fragment (32x16 bf16): lane = {col = lane&15, khalf = lane>>4}.
// Lane holds contiguous K in {kh*16 .. kh*16+15}; B^T rows are W rows.
__device__ __forceinline__ bf16x16 load_fragB(const unsigned short* rowbase, int kh) {
  const unsigned int* p = (const unsigned int*)rowbase;
  u32x8 u;
  int o = kh * 8;
#pragma unroll
  for (int i = 0; i < 8; ++i) u[i] = p[o + i];
  return __builtin_bit_cast(bf16x16, u);
}

__device__ __forceinline__ f32x8 wmma_bf16(bf16x16 a, bf16x16 b, f32x8 c) {
  return __builtin_amdgcn_wmma_f32_16x16x32_bf16(false, a, false, b, (short)0, c,
                                                 false, false);
}

// ===========================================================================
// Generic GEMM: Y[:, col0:col0+N) = X[M,K] @ W[N,K]^T + bias   (fp32 memory)
// ===========================================================================
__global__ __launch_bounds__(256)
void wmma_gemm_xwT(const float* __restrict__ X, int ldx,
                   const float* __restrict__ W, int ldw,
                   const float* __restrict__ bias,
                   float* __restrict__ Y, int ldy, int col0,
                   int M, int N, int K)
{
  __shared__ __align__(16) unsigned short Als[TBM * LDT];
  __shared__ __align__(16) unsigned short Bls[TBN * LDT];

  const int tid  = threadIdx.x;
  const int lane = tid & 31;
  const int wave = tid >> 5;
  const int m0 = blockIdx.y * TBM;
  const int n0 = blockIdx.x * TBN;
  const int wm = (wave >> 1) << 5;   // 0,32,64,96
  const int wn = (wave & 1)  << 5;   // 0,32
  const int lr = lane & 15;
  const int kh = lane >> 4;

  f32x8 acc[2][2];
#pragma unroll
  for (int im = 0; im < 2; ++im)
#pragma unroll
    for (int jn = 0; jn < 2; ++jn) {
      int gn = n0 + wn + jn * 16 + lr;          // C/D layout: col = lane&15
      float bv = (bias && gn < N) ? bias[gn] : 0.0f;
#pragma unroll
      for (int g = 0; g < 8; ++g) acc[im][jn][g] = bv;
    }

  // register double-buffered pipeline: global fetch of chunk k+1 overlaps
  // WMMA on chunk k (LDS), hiding HBM latency behind the matrix pipe.
  float4 ra[4], rb[2];
  fetchA(ra, X, ldx, m0, 0, M, K, tid);
  fetchB(rb, W, ldw, n0, 0, N, K, tid);

  for (int k0 = 0; k0 < K; k0 += TBK) {
    storeA(Als, ra, tid);
    storeB(Bls, rb, tid);
    __syncthreads();

    int kn = k0 + TBK;
    if (kn < K) {
      fetchA(ra, X, ldx, m0, kn, M, K, tid);
      fetchB(rb, W, ldw, n0, kn, N, K, tid);
    }

    bf16x16 a0 = load_fragA(&Als[(wm +      lr) * LDT], kh);
    bf16x16 a1 = load_fragA(&Als[(wm + 16 + lr) * LDT], kh);
    bf16x16 b0 = load_fragB(&Bls[(wn +      lr) * LDT], kh);
    bf16x16 b1 = load_fragB(&Bls[(wn + 16 + lr) * LDT], kh);

    acc[0][0] = wmma_bf16(a0, b0, acc[0][0]);
    acc[0][1] = wmma_bf16(a0, b1, acc[0][1]);
    acc[1][0] = wmma_bf16(a1, b0, acc[1][0]);
    acc[1][1] = wmma_bf16(a1, b1, acc[1][1]);

    __syncthreads();
  }

#pragma unroll
  for (int im = 0; im < 2; ++im)
#pragma unroll
    for (int jn = 0; jn < 2; ++jn) {
      int gn = n0 + wn + jn * 16 + lr;
      if (gn >= N) continue;
      int rb2 = m0 + wm + im * 16 + kh * 8;     // C/D layout: row = g + 8*(lane>>4)
#pragma unroll
      for (int g = 0; g < 8; ++g) {
        int gm = rb2 + g;
        if (gm < M) Y[(size_t)gm * ldy + col0 + gn] = acc[im][jn][g];
      }
    }
}

// ===========================================================================
// Fused single-step BiLSTM: OUT[M,200] (+)= scale * bilstm(X[M,Kd]; Wf,Wb).
// z = X@W^T + b for both directions stays in LDS; gate math (torch order
// i,f,g,o, h0=c0=0: c=sig(i)*tanh(g), h=sig(o)*tanh(c)) fused in-kernel.
// LDS: A 32x232 bf16 (14.5KB) + B 64x40 bf16 (5KB) + z 32x804 f32 (103KB).
// ===========================================================================
__global__ __launch_bounds__(256)
void wmma_bilstm(const float* __restrict__ X, int ldx, int Kd,
                 const float* __restrict__ Wf, const float* __restrict__ bf,
                 const float* __restrict__ Wb, const float* __restrict__ bb,
                 float* __restrict__ OUT, int M, float scale, int accumulate)
{
  __shared__ __align__(16) unsigned short Als[FBM * FLDA];
  __shared__ __align__(16) unsigned short Bls[TBN * LDT];
  __shared__ float zbuf[FBM * ZLD];

  const int tid  = threadIdx.x;
  const int lane = tid & 31;
  const int wave = tid >> 5;
  const int m0 = blockIdx.x * FBM;
  const int mt = (wave >> 2) << 4;   // 0,16
  const int nt = (wave & 3) << 4;    // 0,16,32,48
  const int lr = lane & 15;
  const int kh = lane >> 4;

  // stage A once: rows m0..m0+31, cols zero-padded out to KMAX
  constexpr int CH = KMAX / 4;       // 56 float4 chunks per row
  for (int e = tid; e < FBM * CH; e += 256) {
    int row = e / CH, c4 = (e - row * CH) * 4;
    float4 v;
    v.x = ldg_guard(X, ldx, m0 + row, c4 + 0, M, Kd);
    v.y = ldg_guard(X, ldx, m0 + row, c4 + 1, M, Kd);
    v.z = ldg_guard(X, ldx, m0 + row, c4 + 2, M, Kd);
    v.w = ldg_guard(X, ldx, m0 + row, c4 + 3, M, Kd);
    *(uint2*)&Als[row * FLDA + c4] = cvt4(v);
  }
  __syncthreads();

  for (int side = 0; side < 2; ++side) {
    const float* Wp = side ? Wb : Wf;
    const float* bp = side ? bb : bf;
    for (int n0 = 0; n0 < 400; n0 += TBN) {
      const int cn = n0 + nt + lr;             // gate column (lane's N)
      f32x8 acc;
      float bv = (cn < 400) ? bp[cn] : 0.0f;
#pragma unroll
      for (int g = 0; g < 8; ++g) acc[g] = bv;

      for (int k0 = 0; k0 < Kd; k0 += TBK) {
        float4 rbv[2];
        fetchB(rbv, Wp, Kd, n0, k0, 400, Kd, tid);
        storeB(Bls, rbv, tid);
        __syncthreads();
        bf16x16 a = load_fragA(&Als[(mt + lr) * FLDA + k0], kh);
        bf16x16 b = load_fragB(&Bls[(nt + lr) * LDT], kh);
        acc = wmma_bf16(a, b, acc);
        __syncthreads();
      }
      if (cn < 400) {
        int zc = side * 400 + cn;
#pragma unroll
        for (int g = 0; g < 8; ++g)
          zbuf[(mt + g + 8 * kh) * ZLD + zc] = acc[g];
      }
    }
  }
  __syncthreads();

  // fused gate epilogue: only h [32,200] leaves the WGP
  for (int t = tid; t < FBM * 200; t += 256) {
    int m = t / 200, j = t - m * 200;
    int dir = j / 100, jj = j - dir * 100;
    const float* z = &zbuf[m * ZLD + dir * 400];
    float i = z[jj], g = z[200 + jj], o = z[300 + jj];
    float c = (1.0f / (1.0f + __expf(-i))) * tanhf(g);
    float h = (1.0f / (1.0f + __expf(-o))) * tanhf(c);
    int gm = m0 + m;
    if (gm < M) {
      float* d = OUT + (size_t)gm * 200 + j;
      *d = accumulate ? (*d + scale * h) : (scale * h);
    }
  }
}

// mean over K=20 neighbors: X [Bn*20, 200] -> O [Bn, 200]
__global__ void mean_over_k(const float* __restrict__ X, float* __restrict__ O, int Bn)
{
  int t = blockIdx.x * blockDim.x + threadIdx.x;
  if (t >= Bn * 200) return;
  int b = t / 200, j = t - b * 200;
  float s = 0.0f;
#pragma unroll
  for (int k = 0; k < 20; ++k) s += X[((size_t)b * 20 + k) * 200 + j];
  O[(size_t)b * 200 + j] = s * (1.0f / 20.0f);
}

// attention (leaky_relu slope 0.01) + softmax over 3 + combine + sigmoid head
__global__ void attn_out(const float* __restrict__ CS,
                         const float* __restrict__ NA, const float* __restrict__ PA,
                         const float* __restrict__ UA,
                         const float* __restrict__ natt, const float* __restrict__ patt,
                         const float* __restrict__ uatt,
                         const float* __restrict__ outW, const float* __restrict__ outB,
                         float* __restrict__ out, int Bn)
{
  int b = blockIdx.x * blockDim.x + threadIdx.x;
  if (b >= Bn) return;
  const float* atts[3] = {natt, patt, uatt};
  const float* aggs[3] = {NA, PA, UA};
  float s[3];
#pragma unroll
  for (int t = 0; t < 3; ++t) {
    float acc = 0.0f;
    for (int j = 0; j < 200; ++j) acc += CS[(size_t)b * 200 + j] * atts[t][j];
    for (int j = 0; j < 200; ++j) acc += aggs[t][(size_t)b * 200 + j] * atts[t][200 + j];
    s[t] = acc > 0.0f ? acc : 0.01f * acc;
  }
  float mx = fmaxf(s[0], fmaxf(s[1], s[2]));
  float e0 = __expf(s[0] - mx), e1 = __expf(s[1] - mx), e2 = __expf(s[2] - mx);
  float inv = 1.0f / (e0 + e1 + e2);
  float a0 = e0 * inv, a1 = e1 * inv, a2 = e2 * inv;
  float acc = outB[0];
  for (int j = 0; j < 200; ++j) {
    float comb = a0 * NA[(size_t)b * 200 + j] + a1 * PA[(size_t)b * 200 + j]
               + a2 * UA[(size_t)b * 200 + j];
    acc += comb * outW[j];
  }
  out[b] = 1.0f / (1.0f + __expf(-acc));
}

// ---------------------------------------------------------------------------
// Input index map. d_in[0..9]: arrays in setup_inputs() insertion order.
// d_in[10..70]: params flattened jax-pytree style (dict keys sorted; within a
// linear: W,b; within a bilstm: Wb,Wf,bb,bf).
// ---------------------------------------------------------------------------
enum {
  IN_SELF_TITLE = 0, IN_SELF_CONTENT, IN_SELF_IMAGE,
  IN_NB_TITLE, IN_NB_CONTENT, IN_NB_IMAGE,
  IN_POST_TEXT, IN_POST_OTHER /*unused*/, IN_USER_TEXT, IN_USER_OTHER,
  P_LIN_IMAGE_W, P_LIN_IMAGE_B,
  P_LIN_OTHER_W, P_LIN_OTHER_B,            // unused by forward
  P_LIN_OU_W, P_LIN_OU_B,
  P_LIN_TEXT_W, P_LIN_TEXT_B,
  P_L_IMG_WB, P_L_IMG_WF, P_L_IMG_BB, P_L_IMG_BF,
  P_L_NC_WB, P_L_NC_WF, P_L_NC_BB, P_L_NC_BF,
  P_L_NT_WB, P_L_NT_WF, P_L_NT_BB, P_L_NT_BF,
  P_L_OU_WB, P_L_OU_WF, P_L_OU_BB, P_L_OU_BF,
  P_L_PT_WB, P_L_PT_WF, P_L_PT_BB, P_L_PT_BF,
  P_L_UT_WB, P_L_UT_WF, P_L_UT_BB, P_L_UT_BF,
  P_N_ATT, P_N_INIT_W, P_N_INIT_B,
  P_N_LSTM_WB, P_N_LSTM_WF, P_N_LSTM_BB, P_N_LSTM_BF,
  P_N_OUT_W, P_N_OUT_B,
  P_OUT_LIN_W, P_OUT_LIN_B,
  P_P_ATT, P_P_INIT_W, P_P_INIT_B,
  P_P_LSTM_WB, P_P_LSTM_WF, P_P_LSTM_BB, P_P_LSTM_BF,
  P_P_OUT_W, P_P_OUT_B,
  P_U_ATT, P_U_INIT_W, P_U_INIT_B,
  P_U_LSTM_WB, P_U_LSTM_WF, P_U_LSTM_BB, P_U_LSTM_BF,
  P_U_OUT_W, P_U_OUT_B
};

static inline void gemm(hipStream_t s, const float* X, int ldx,
                        const float* W, int ldw, const float* bias,
                        float* Y, int ldy, int col0, int M, int N, int K)
{
  dim3 grid((N + TBN - 1) / TBN, (M + TBM - 1) / TBM);
  wmma_gemm_xwT<<<grid, 256, 0, s>>>(X, ldx, W, ldw, bias, Y, ldy, col0, M, N, K);
}

static inline void bilstm_f(hipStream_t s, const float* X, int ldx, int Kd,
                            const float* wf, const float* bf,
                            const float* wb, const float* bb,
                            float* O, int M, float scale, int acc)
{
  wmma_bilstm<<<(M + FBM - 1) / FBM, 256, 0, s>>>(X, ldx, Kd, wf, bf, wb, bb,
                                                  O, M, scale, acc);
}

extern "C" void kernel_launch(void* const* d_in, const int* in_sizes, int n_in,
                              void* d_out, int out_size, void* d_ws, size_t ws_size,
                              hipStream_t stream)
{
  (void)in_sizes; (void)n_in; (void)out_size; (void)ws_size;
  const int B = 1024, K = 20, M = B * K;   // 20480 neighbor rows

  auto F = [&](int i) -> const float* { return (const float*)d_in[i]; };

  // ---- workspace carve-up (fp32, ~105 MB; buffers reused sequentially) ----
  float* ws  = (float*)d_ws;
  float* LIN = ws;                      // [20480, 200] projection outputs
  float* CN  = LIN + (size_t)M * 200;   // content_news (neighbors)
  float* CP  = CN  + (size_t)M * 200;   // content_post
  float* CU  = CP  + (size_t)M * 200;   // content_user
  float* AG  = CU  + (size_t)M * 200;   // agg scratch
  float* AG2 = AG  + (size_t)M * 200;   // agg scratch 2
  float* LOU = AG2 + (size_t)M * 200;   // [20480, 32] other_user lin out
  float* CS  = LOU + (size_t)M * 32;    // [1024, 200] self content
  float* NA  = CS  + (size_t)B * 200;
  float* PA  = NA  + (size_t)B * 200;
  float* UA  = PA  + (size_t)B * 200;

  const float* ltW = F(P_LIN_TEXT_W);  const float* ltB = F(P_LIN_TEXT_B);
  const float* liW = F(P_LIN_IMAGE_W); const float* liB = F(P_LIN_IMAGE_B);

  // one content branch = projection GEMM + fused BiLSTM (gates in LDS)
  auto branch = [&](const float* x, int ldx, int rows,
                    const float* lw, const float* lb, int lk, int ln,
                    const float* wf, const float* bf,
                    const float* wb, const float* bb,
                    float* linbuf, float* hout, float scale, int acc) {
    gemm(stream, x, ldx, lw, lk, lb, linbuf, ln, 0, rows, ln, lk);
    bilstm_f(stream, linbuf, ln, ln, wf, bf, wb, bb, hout, rows, scale, acc);
  };

  // ---- self content_news: (title + content + image)/3 -> CS [1024,200] ----
  branch(F(IN_SELF_TITLE),   768, B, ltW, ltB, 768, 200,
         F(P_L_NT_WF), F(P_L_NT_BF), F(P_L_NT_WB), F(P_L_NT_BB), LIN, CS, 1.f/3.f, 0);
  branch(F(IN_SELF_CONTENT), 768, B, ltW, ltB, 768, 200,
         F(P_L_NC_WF), F(P_L_NC_BF), F(P_L_NC_WB), F(P_L_NC_BB), LIN, CS, 1.f/3.f, 1);
  branch(F(IN_SELF_IMAGE),   512, B, liW, liB, 512, 200,
         F(P_L_IMG_WF), F(P_L_IMG_BF), F(P_L_IMG_WB), F(P_L_IMG_BB), LIN, CS, 1.f/3.f, 1);

  // ---- neighbor content_news -> CN [20480,200] ----
  branch(F(IN_NB_TITLE),   768, M, ltW, ltB, 768, 200,
         F(P_L_NT_WF), F(P_L_NT_BF), F(P_L_NT_WB), F(P_L_NT_BB), LIN, CN, 1.f/3.f, 0);
  branch(F(IN_NB_CONTENT), 768, M, ltW, ltB, 768, 200,
         F(P_L_NC_WF), F(P_L_NC_BF), F(P_L_NC_WB), F(P_L_NC_BB), LIN, CN, 1.f/3.f, 1);
  branch(F(IN_NB_IMAGE),   512, M, liW, liB, 512, 200,
         F(P_L_IMG_WF), F(P_L_IMG_BF), F(P_L_IMG_WB), F(P_L_IMG_BB), LIN, CN, 1.f/3.f, 1);

  // ---- content_post (text only) -> CP ----
  branch(F(IN_POST_TEXT), 768, M, ltW, ltB, 768, 200,
         F(P_L_PT_WF), F(P_L_PT_BF), F(P_L_PT_WB), F(P_L_PT_BB), LIN, CP, 1.0f, 0);

  // ---- content_user: (text + other_user)/2 -> CU ----
  branch(F(IN_USER_TEXT), 768, M, ltW, ltB, 768, 200,
         F(P_L_UT_WF), F(P_L_UT_BF), F(P_L_UT_WB), F(P_L_UT_BB), LIN, CU, 0.5f, 0);
  branch(F(IN_USER_OTHER), 11, M, F(P_LIN_OU_W), F(P_LIN_OU_B), 11, 32,
         F(P_L_OU_WF), F(P_L_OU_BF), F(P_L_OU_WB), F(P_L_OU_BB), LOU, CU, 0.5f, 1);

  // ---- SameType_Agg_Bi_RNN per type ----
  auto agg = [&](const float* Ct,
                 const float* iw, const float* ib,
                 const float* wf, const float* bf,
                 const float* wb, const float* bb,
                 const float* ow, const float* ob, float* outBx200) {
    gemm(stream, Ct, 200, iw, 200, ib, AG, 200, 0, M, 200, 200);
    bilstm_f(stream, AG, 200, 200, wf, bf, wb, bb, AG2, M, 1.0f, 0);
    gemm(stream, AG2, 200, ow, 200, ob, AG, 200, 0, M, 200, 200);
    mean_over_k<<<(B * 200 + 255) / 256, 256, 0, stream>>>(AG, outBx200, B);
  };

  agg(CN, F(P_N_INIT_W), F(P_N_INIT_B), F(P_N_LSTM_WF), F(P_N_LSTM_BF),
      F(P_N_LSTM_WB), F(P_N_LSTM_BB), F(P_N_OUT_W), F(P_N_OUT_B), NA);
  agg(CP, F(P_P_INIT_W), F(P_P_INIT_B), F(P_P_LSTM_WF), F(P_P_LSTM_BF),
      F(P_P_LSTM_WB), F(P_P_LSTM_BB), F(P_P_OUT_W), F(P_P_OUT_B), PA);
  agg(CU, F(P_U_INIT_W), F(P_U_INIT_B), F(P_U_LSTM_WF), F(P_U_LSTM_BF),
      F(P_U_LSTM_WB), F(P_U_LSTM_BB), F(P_U_OUT_W), F(P_U_OUT_B), UA);

  // ---- attention + head -> d_out [1024] ----
  attn_out<<<(B + 255) / 256, 256, 0, stream>>>(
      CS, NA, PA, UA, F(P_N_ATT), F(P_P_ATT), F(P_U_ATT),
      F(P_OUT_LIN_W), F(P_OUT_LIN_B), (float*)d_out, B);
}
